// NoiseRobustAttractorLayer_18734647345648
// MI455X (gfx1250) — matrix operand order repaired and checked by reading, not compiled, not run on hardware
//
#include <hip/hip_runtime.h>
#include <math.h>

typedef __attribute__((ext_vector_type(16))) _Float16 v16h;
typedef __attribute__((ext_vector_type(8)))  _Float16 v8h;
typedef __attribute__((ext_vector_type(8)))  float    v8f;

namespace {
constexpr int kIn   = 512;
constexpr int kHid  = 512;
constexpr int kNH   = 4;
constexpr int kHD   = 128;
constexpr int kTHD  = 256;   // 2*HD
constexpr int kNA   = 8;
constexpr int kM    = 64;    // token tile per workgroup
constexpr int kThr  = 256;   // 8 waves (wave32)

// LDS row strides (in halfs). 520*2B=1040B=260 dwords, 260%64=4 -> conflict-free
// 16B A-frag reads (banks 4L..4L+3 distinct across 16 lanes). 264 -> 132dw%64=4.
// 136 -> 68dw%64=4.
constexpr int ldXN = 520;
constexpr int ldH  = 264;
constexpr int ldST = 136;
constexpr int ldCB = 520;
constexpr int ldO  = 520;    // s_o aliases s_xn region, same stride

// dynamic smem byte offsets (all 16B aligned)
constexpr int smXN  = 0;
constexpr int smCB  = smXN + kM * ldXN * 2;      // 66560
constexpr int smH   = smCB + kM * ldCB * 2;      // 133120 (33792 bytes)
constexpr int smST  = smH;                       // alias inside s_h (17408 <= 33792)
constexpr int smO   = smXN;                      // alias over s_xn after head loop
constexpr int smATT = smH + kM * ldH * 2;        // 166912 (4096)
constexpr int smGT  = smATT + kNA * kHD * 4;     // 171008 (64*4*4 = 1024)
constexpr int smMU  = smGT + kM * kNH * 4;       // 172032 (256)
constexpr int smRS  = smMU + kM * 4;             // 172288 (256)
constexpr int smTotal = smRS + kM * 4;           // 172544 bytes (< 320KB/WGP)

// packed f16 weight offsets inside d_ws (in halfs)
constexpr int wsHW1 = 0;                                   // 4*512*256
constexpr int wsHW2 = wsHW1 + kNH * kIn * kTHD;            // 4*256*128
constexpr int wsDYN = wsHW2 + kNH * kTHD * kHD;            // 4*128*128
constexpr int wsOW1 = wsDYN + kNH * kHD * kHD;             // 512*512
constexpr int wsOW2 = wsOW1 + kHid * kIn;                  // 512*512
constexpr int wsHalfTotal = wsOW2 + kHid * kIn;            // 1,245,184 halfs
} // namespace

// ---------- prep kernels ----------

// Repack a K x N (row-major, k-major) f32 matrix into WMMA B-fragment order:
// frag index = nt*(K/32)+kt ; within frag: lane (0..31) -> N = nt*16 + lane%16,
// lane holds K = kt*32 + (lane/16)*8 + [0,8) ++ kt*32 + 16 + (lane/16)*8 + [0,8).
__global__ void pack_b_frags(const float* __restrict__ src, _Float16* __restrict__ dst,
                             int K, int N) {
  int idx = blockIdx.x * blockDim.x + threadIdx.x;
  if (idx >= K * N) return;
  int fragIdx = idx >> 9;
  int within  = idx & 511;
  int lane    = within >> 4;
  int t       = within & 15;
  int KT = K >> 5;
  int nt = fragIdx / KT;
  int kt = fragIdx - nt * KT;
  int n = (nt << 4) + (lane & 15);
  int k = (kt << 5) + ((lane >> 4) << 3) + (t & 7) + ((t >> 3) << 4);
  dst[idx] = (_Float16)src[k * N + n];
}

__global__ void compute_a2(const float* __restrict__ attr, float* __restrict__ a2) {
  int i = threadIdx.x;
  if (i < kNH * kNA) {
    const float* p = attr + i * kHD;
    float s = 0.f;
    for (int d = 0; d < kHD; ++d) s += p[d] * p[d];
    a2[i] = s;
  }
}

// ---------- fused kernel helpers ----------

__device__ __forceinline__ v8f zero8() {
  v8f z;
#pragma unroll
  for (int i = 0; i < 8; ++i) z[i] = 0.f;
  return z;
}

// A-fragment from LDS per documented 16-bit A layout:
// lane L: row = L%16; halfs = K in [b,b+8) ++ [b+16,b+24), b = 8*(L/16).
__device__ __forceinline__ v16h load_a(const _Float16* base) {
  v8h c0 = *(const v8h*)base;
  v8h c1 = *(const v8h*)(base + 16);
  return __builtin_shufflevector(c0, c1, 0,1,2,3,4,5,6,7,8,9,10,11,12,13,14,15);
}

__device__ __forceinline__ v8f wmma16(v16h a, v16h b, v8f c) {
  return __builtin_amdgcn_wmma_f32_16x16x32_f16(false, a, false, b, (short)0, c,
                                                false, false);
}

__device__ __forceinline__ float qsum(float v) {
  v += __shfl_xor(v, 1, 32);
  v += __shfl_xor(v, 2, 32);
  return v;
}

__device__ __forceinline__ float gelu_exact(float v) {
  return 0.5f * v * (1.f + erff(v * 0.70710678118654752f));
}

__global__ __launch_bounds__(kThr, 1)
void fused_attractor(const float* __restrict__ x,
                     const float* __restrict__ norm_g, const float* __restrict__ norm_b,
                     const float* __restrict__ hb1,
                     const float* __restrict__ hln_g, const float* __restrict__ hln_b,
                     const float* __restrict__ hb2,
                     const float* __restrict__ attractors,
                     const float* __restrict__ gW, const float* __restrict__ gb,
                     const float* __restrict__ ob1,
                     const float* __restrict__ oln_g, const float* __restrict__ oln_b,
                     const float* __restrict__ ob2,
                     const _Float16* __restrict__ wf,
                     const float* __restrict__ a2v,
                     float* __restrict__ out) {
  extern __shared__ __align__(16) unsigned char smem[];
  _Float16* s_xn = (_Float16*)(smem + smXN);
  _Float16* s_cb = (_Float16*)(smem + smCB);
  _Float16* s_h  = (_Float16*)(smem + smH);
  _Float16* s_st = (_Float16*)(smem + smST);   // alias of s_h region
  _Float16* s_o  = (_Float16*)(smem + smO);    // alias of s_xn region
  float*    s_att  = (float*)(smem + smATT);
  float*    s_gate = (float*)(smem + smGT);
  float*    s_mu   = (float*)(smem + smMU);
  float*    s_rs   = (float*)(smem + smRS);

  const int tid  = threadIdx.x;
  const int lane = tid & 31;
  const int w    = tid >> 5;
  const int ms   = w >> 1;   // M strip (16 rows)
  const int nhv  = w & 1;    // N half
  const long rowG0 = (long)blockIdx.x * kM;

  // ---- phase 0: LayerNorm(x) -> s_xn (f16); per-row mean/rstd; all 4 gates ----
  {
    const int r = tid >> 2, q = tid & 3;  // 4 threads/row, 128 cols each
    const float4* xr = (const float4*)(x + (rowG0 + r) * kIn + q * 128);
    float4 buf[32];
    float s = 0.f, ss = 0.f;
#pragma unroll
    for (int i = 0; i < 32; ++i) {
      buf[i] = xr[i];
      s  += buf[i].x + buf[i].y + buf[i].z + buf[i].w;
      ss += buf[i].x * buf[i].x + buf[i].y * buf[i].y +
            buf[i].z * buf[i].z + buf[i].w * buf[i].w;
    }
    s = qsum(s); ss = qsum(ss);
    float mean = s * (1.f / 512.f);
    float rstd = rsqrtf(ss * (1.f / 512.f) - mean * mean + 1e-5f);
    if (q == 0) { s_mu[r] = mean; s_rs[r] = rstd; }
    _Float16* dst = s_xn + r * ldXN + q * 128;
    float gdot[kNH] = {0.f, 0.f, 0.f, 0.f};
#pragma unroll
    for (int i = 0; i < 32; ++i) {
      int c = q * 128 + i * 4;
      float xv0 = (buf[i].x - mean) * rstd * norm_g[c+0] + norm_b[c+0];
      float xv1 = (buf[i].y - mean) * rstd * norm_g[c+1] + norm_b[c+1];
      float xv2 = (buf[i].z - mean) * rstd * norm_g[c+2] + norm_b[c+2];
      float xv3 = (buf[i].w - mean) * rstd * norm_g[c+3] + norm_b[c+3];
      dst[i*4+0] = (_Float16)xv0;
      dst[i*4+1] = (_Float16)xv1;
      dst[i*4+2] = (_Float16)xv2;
      dst[i*4+3] = (_Float16)xv3;
#pragma unroll
      for (int h = 0; h < kNH; ++h) {
        float4 wv = ((const float4*)(gW + h * kIn + q * 128))[i];
        gdot[h] += xv0 * wv.x + xv1 * wv.y + xv2 * wv.z + xv3 * wv.w;
      }
    }
#pragma unroll
    for (int h = 0; h < kNH; ++h) {
      float g = qsum(gdot[h]);
      if (q == 0) s_gate[r * kNH + h] = 1.f / (1.f + expf(-(g + gb[h])));
    }
  }
  __syncthreads();

  const int arow16 = ms * 16 + (lane & 15);
  const int aks    = (lane >> 4) << 3;
  const int r0c    = ms * 16 + ((lane >> 4) << 3);

  // ---- per-head pipeline ----
  for (int h = 0; h < kNH; ++h) {
    // GEMM1: [64,512] x [512,256] -> s_h (wave: 16x128 = 8 tiles)
    {
      v8f acc[8];
#pragma unroll
      for (int j = 0; j < 8; ++j) acc[j] = zero8();
      const _Float16* wb = wf + wsHW1 + h * (kIn * kTHD);
      __builtin_prefetch(wb + ((nhv * 8 * 16) << 9) + (lane << 4), 0, 1);
#pragma unroll 4
      for (int kt = 0; kt < 16; ++kt) {
        v16h bfr[8];
#pragma unroll
        for (int j = 0; j < 8; ++j)
          bfr[j] = *(const v16h*)(wb + (((nhv * 8 + j) * 16 + kt) << 9) + (lane << 4));
        v16h a = load_a(s_xn + arow16 * ldXN + kt * 32 + aks);
#pragma unroll
        for (int j = 0; j < 8; ++j) acc[j] = wmma16(a, bfr[j], acc[j]);
      }
#pragma unroll
      for (int j = 0; j < 8; ++j) {
        int col = nhv * 128 + j * 16 + (lane & 15);
#pragma unroll
        for (int i = 0; i < 8; ++i)
          s_h[(r0c + i) * ldH + col] = (_Float16)acc[j][i];
      }
    }
    __syncthreads();

    // row phase A: +hb1, LN(256), GELU, in place on s_h
    {
      const int r = tid >> 2, q = tid & 3;  // 64 cols each
      float hv[64]; float s = 0.f, ss = 0.f;
      const float* b1 = hb1 + h * kTHD + q * 64;
      _Float16* hp = s_h + r * ldH + q * 64;
#pragma unroll
      for (int c = 0; c < 64; ++c) {
        float v = (float)hp[c] + b1[c];
        hv[c] = v; s += v; ss += v * v;
      }
      s = qsum(s); ss = qsum(ss);
      float mean = s * (1.f / 256.f);
      float rstd = rsqrtf(ss * (1.f / 256.f) - mean * mean + 1e-5f);
      const float* lg = hln_g + h * kTHD + q * 64;
      const float* lb = hln_b + h * kTHD + q * 64;
#pragma unroll
      for (int c = 0; c < 64; ++c) {
        float v = (hv[c] - mean) * rstd * lg[c] + lb[c];
        hp[c] = (_Float16)gelu_exact(v);
      }
    }
    __syncthreads();

    // GEMM2: [64,256] x [256,128] + hb2 -> s_st (aliased over s_h, so
    // k-loop first, barrier, then writeback)
    v8f acc2[4];
    {
#pragma unroll
      for (int j = 0; j < 4; ++j) acc2[j] = zero8();
      const _Float16* wb = wf + wsHW2 + h * (kTHD * kHD);
      const int nt0 = nhv * 4;
#pragma unroll 2
      for (int kt = 0; kt < 8; ++kt) {
        v16h bfr[4];
#pragma unroll
        for (int j = 0; j < 4; ++j)
          bfr[j] = *(const v16h*)(wb + (((nt0 + j) * 8 + kt) << 9) + (lane << 4));
        v16h a = load_a(s_h + arow16 * ldH + kt * 32 + aks);
#pragma unroll
        for (int j = 0; j < 4; ++j) acc2[j] = wmma16(a, bfr[j], acc2[j]);
      }
    }
    __syncthreads();  // all s_h A-reads done before overwriting region with s_st
    {
      const int nt0 = nhv * 4;
#pragma unroll
      for (int j = 0; j < 4; ++j) {
        int col = (nt0 + j) * 16 + (lane & 15);
        float bias = hb2[h * kHD + col];
#pragma unroll
        for (int i = 0; i < 8; ++i)
          s_st[(r0c + i) * ldST + col] = (_Float16)(acc2[j][i] + bias);
      }
    }
    // stage attractors for this head (prev head's readers finished last barrier)
    for (int i2 = tid; i2 < kNA * kHD; i2 += kThr)
      s_att[i2] = attractors[h * kNA * kHD + i2];
    __syncthreads();

    // dynamics GEMM k-loop: [64,128] x [128,128] (reads s_st; no writes yet)
    v8f accD[4];
    {
#pragma unroll
      for (int j = 0; j < 4; ++j) accD[j] = zero8();
      const _Float16* wb = wf + wsDYN + h * (kHD * kHD);
      const int nt0 = nhv * 4;
#pragma unroll
      for (int kt = 0; kt < 4; ++kt) {
        v16h bfr[4];
#pragma unroll
        for (int j = 0; j < 4; ++j)
          bfr[j] = *(const v16h*)(wb + (((nt0 + j) * 4 + kt) << 9) + (lane << 4));
        v16h a = load_a(s_st + arow16 * ldST + kt * 32 + aks);
#pragma unroll
        for (int j = 0; j < 4; ++j) accD[j] = wmma16(a, bfr[j], accD[j]);
      }
    }

    // row phase B: cdist/softmax/attractor partial combine -> s_cb
    // partial = (1-gs)*state + gs*attr_infl ; dyn writeback later adds gs*tanh(dyn)
    {
      const int r = tid >> 2, q = tid & 3, d0 = q * 32;
      float sv[32];
      float s2 = 0.f;
      float sa[8] = {0.f, 0.f, 0.f, 0.f, 0.f, 0.f, 0.f, 0.f};
#pragma unroll
      for (int d = 0; d < 32; ++d) {
        float v = (float)s_st[r * ldST + d0 + d];
        sv[d] = v; s2 += v * v;
#pragma unroll
        for (int a = 0; a < kNA; ++a) sa[a] += v * s_att[a * kHD + d0 + d];
      }
      s2 = qsum(s2);
#pragma unroll
      for (int a = 0; a < kNA; ++a) sa[a] = qsum(sa[a]);
      float lgt[8]; float mx = -3.0e38f;
#pragma unroll
      for (int a = 0; a < kNA; ++a) {
        float d2 = fmaxf(s2 + a2v[h * kNA + a] - 2.f * sa[a], 0.f);
        lgt[a] = sqrtf(d2) * (-0.08838834764831845f);  // -1/sqrt(128)
        mx = fmaxf(mx, lgt[a]);
      }
      float se = 0.f;
#pragma unroll
      for (int a = 0; a < kNA; ++a) { lgt[a] = expf(lgt[a] - mx); se += lgt[a]; }
      float inv = 1.f / se;
      float gs = 0.1f * s_gate[r * kNH + h];
      _Float16* cb = s_cb + r * ldCB + h * kHD + d0;
#pragma unroll
      for (int d = 0; d < 32; ++d) {
        float ai = 0.f;
#pragma unroll
        for (int a = 0; a < kNA; ++a) ai += lgt[a] * s_att[a * kHD + d0 + d];
        ai *= inv;
        cb[d] = (_Float16)(sv[d] + gs * (ai - sv[d]));
      }
    }
    __syncthreads();

    // dynamics writeback: s_cb += gate*0.1*tanh(dyn)
    {
      const int nt0 = nhv * 4;
#pragma unroll
      for (int j = 0; j < 4; ++j) {
        int col = (nt0 + j) * 16 + (lane & 15);
#pragma unroll
        for (int i = 0; i < 8; ++i) {
          int r = r0c + i;
          float gs = 0.1f * s_gate[r * kNH + h];
          _Float16* cp = s_cb + r * ldCB + h * kHD + col;
          *cp = (_Float16)((float)*cp + gs * tanhf(accD[j][i]));
        }
      }
    }
    __syncthreads();
  } // head loop

  // ---- GEMM3: [64,512] x [512,512] -> s_o (= s_xn region; two N passes) ----
  for (int p = 0; p < 2; ++p) {
    v8f acc[8];
#pragma unroll
    for (int j = 0; j < 8; ++j) acc[j] = zero8();
    const _Float16* wb = wf + wsOW1;
    __builtin_prefetch(wb + (((p * 16 + nhv * 8) * 16) << 9) + (lane << 4), 0, 1);
#pragma unroll 2
    for (int kt = 0; kt < 16; ++kt) {
      v16h bfr[8];
#pragma unroll
      for (int j = 0; j < 8; ++j)
        bfr[j] = *(const v16h*)(wb + (((p * 16 + nhv * 8 + j) * 16 + kt) << 9) + (lane << 4));
      v16h a = load_a(s_cb + arow16 * ldCB + kt * 32 + aks);
#pragma unroll
      for (int j = 0; j < 8; ++j) acc[j] = wmma16(a, bfr[j], acc[j]);
    }
#pragma unroll
    for (int j = 0; j < 8; ++j) {
      int col = (p * 16 + nhv * 8 + j) * 16 + (lane & 15);
#pragma unroll
      for (int i = 0; i < 8; ++i)
        s_o[(r0c + i) * ldO + col] = (_Float16)acc[j][i];
    }
  }
  __syncthreads();

  // ---- row phase C: +ob1, LN(512), GELU in place on s_o ----
  {
    const int r = tid >> 2, q = tid & 3;  // 128 cols each
    float ov[128]; float s = 0.f, ss = 0.f;
    _Float16* op = s_o + r * ldO + q * 128;
    const float* b1 = ob1 + q * 128;
#pragma unroll
    for (int c = 0; c < 128; ++c) {
      float v = (float)op[c] + b1[c];
      ov[c] = v; s += v; ss += v * v;
    }
    s = qsum(s); ss = qsum(ss);
    float mean = s * (1.f / 512.f);
    float rstd = rsqrtf(ss * (1.f / 512.f) - mean * mean + 1e-5f);
    const float* lg = oln_g + q * 128;
    const float* lb = oln_b + q * 128;
#pragma unroll
    for (int c = 0; c < 128; ++c) {
      float v = (ov[c] - mean) * rstd * lg[c] + lb[c];
      op[c] = (_Float16)gelu_exact(v);
    }
  }
  __syncthreads();

  // ---- GEMM4: [64,512] x [512,512] + ob2 + residual xn -> global out ----
  for (int p = 0; p < 2; ++p) {
    v8f acc[8];
#pragma unroll
    for (int j = 0; j < 8; ++j) acc[j] = zero8();
    const _Float16* wb = wf + wsOW2;
    __builtin_prefetch(wb + (((p * 16 + nhv * 8) * 16) << 9) + (lane << 4), 0, 1);
#pragma unroll 2
    for (int kt = 0; kt < 16; ++kt) {
      v16h bfr[8];
#pragma unroll
      for (int j = 0; j < 8; ++j)
        bfr[j] = *(const v16h*)(wb + (((p * 16 + nhv * 8 + j) * 16 + kt) << 9) + (lane << 4));
      v16h a = load_a(s_o + arow16 * ldO + kt * 32 + aks);
#pragma unroll
      for (int j = 0; j < 8; ++j) acc[j] = wmma16(a, bfr[j], acc[j]);
    }
#pragma unroll
    for (int j = 0; j < 8; ++j) {
      int col = (p * 16 + nhv * 8 + j) * 16 + (lane & 15);
      float b2 = ob2[col];
      float ng = norm_g[col], nb = norm_b[col];
#pragma unroll
      for (int i = 0; i < 8; ++i) {
        int r = r0c + i;
        long gi = (rowG0 + r) * kIn + col;
        float xn = (x[gi] - s_mu[r]) * s_rs[r] * ng + nb;  // f32 residual (x in L2)
        out[gi] = xn + acc[j][i] + b2;
      }
    }
  }
}

// ---------- launch ----------

extern "C" void kernel_launch(void* const* d_in, const int* in_sizes, int n_in,
                              void* d_out, int out_size, void* d_ws, size_t ws_size,
                              hipStream_t stream) {
  (void)in_sizes; (void)n_in; (void)out_size; (void)ws_size;
  const float* x      = (const float*)d_in[0];
  const float* norm_g = (const float*)d_in[1];
  const float* norm_b = (const float*)d_in[2];
  const float* hW1    = (const float*)d_in[3];
  const float* hb1    = (const float*)d_in[4];
  const float* hln_g  = (const float*)d_in[5];
  const float* hln_b  = (const float*)d_in[6];
  const float* hW2    = (const float*)d_in[7];
  const float* hb2    = (const float*)d_in[8];
  const float* attr   = (const float*)d_in[9];
  const float* dyn    = (const float*)d_in[10];
  const float* gW     = (const float*)d_in[11];
  const float* gb     = (const float*)d_in[12];
  const float* oW1    = (const float*)d_in[13];
  const float* ob1    = (const float*)d_in[14];
  const float* oln_g  = (const float*)d_in[15];
  const float* oln_b  = (const float*)d_in[16];
  const float* oW2    = (const float*)d_in[17];
  const float* ob2    = (const float*)d_in[18];
  float* out = (float*)d_out;

  _Float16* wsh = (_Float16*)d_ws;
  float* a2v = (float*)((char*)d_ws + (size_t)wsHalfTotal * 2);

  // weight repack (hot in L2 thereafter)
  for (int h = 0; h < kNH; ++h) {
    pack_b_frags<<<(kIn * kTHD + 255) / 256, 256, 0, stream>>>(
        hW1 + (size_t)h * kIn * kTHD, wsh + wsHW1 + h * kIn * kTHD, kIn, kTHD);
    pack_b_frags<<<(kTHD * kHD + 255) / 256, 256, 0, stream>>>(
        hW2 + (size_t)h * kTHD * kHD, wsh + wsHW2 + h * kTHD * kHD, kTHD, kHD);
    pack_b_frags<<<(kHD * kHD + 255) / 256, 256, 0, stream>>>(
        dyn + (size_t)h * kHD * kHD, wsh + wsDYN + h * kHD * kHD, kHD, kHD);
  }
  pack_b_frags<<<(kHid * kIn + 255) / 256, 256, 0, stream>>>(oW1, wsh + wsOW1, kHid, kIn);
  pack_b_frags<<<(kHid * kIn + 255) / 256, 256, 0, stream>>>(oW2, wsh + wsOW2, kHid, kIn);
  compute_a2<<<1, 32, 0, stream>>>(attr, a2v);

  const int tokens = 16 * 4096;
  fused_attractor<<<tokens / kM, kThr, smTotal, stream>>>(
      x, norm_g, norm_b, hb1, hln_g, hln_b, hb2, attr, gW, gb,
      ob1, oln_g, oln_b, ob2, wsh, a2v, out);
}
